// Seq2Seq_71683004170247
// MI455X (gfx1250) — matrix operand order repaired and checked by reading, not compiled, not run on hardware
//
#include <hip/hip_runtime.h>
#include <hip/hip_bf16.h>
#include <math.h>

// ---------------------------------------------------------------------------
// Seq2Seq (bi-GRU encoder + Bahdanau attention decoder) for MI455X (gfx1250)
//
// Roofline: 201 GFLOP dominated by 32x logits GEMM [64x1536]x[1536x32000].
// W_lsm packed once to f16 (98MB) -> L2-resident (192MB L2) across all 32
// decoder steps. GEMM waves register-block 64(M)x32(N) so every packed B
// fragment is fetched from L2 exactly once (4x L2 traffic reduction vs
// per-M-tile waves). All GEMMs use v_wmma_f32_16x16x32_f16 on pre-packed
// fragments (ISA 7.12.2 layouts). Bahdanau Uh precomputed once.
// ---------------------------------------------------------------------------

typedef __attribute__((ext_vector_type(16))) _Float16 v16h;
typedef __attribute__((ext_vector_type(8)))  float    v8f;

#define TX 64
#define BN 64
#define TY 32
#define EE 256
#define HH 512
#define KY 32000

// ---------------- fragment packing (ISA 7.12.2 layouts) --------------------
// A (16x32 f16): lane = half(=l>>4)*16 + m(=l&15); slot i -> v=i>>1, hh=i&1,
//   k_local = (v>>2)*16 + half*8 + (v&3)*2 + hh
__global__ void k_pack_a(const float* __restrict__ A, int ld, int M, int K,
                         _Float16* __restrict__ out) {
  int Kt = K >> 5;
  int tid = blockIdx.x * blockDim.x + threadIdx.x;
  int total = (M >> 4) * Kt * 32;
  if (tid >= total) return;
  int lane = tid & 31;
  int tile = tid >> 5;
  int kt = tile % Kt, mt = tile / Kt;
  int half = lane >> 4, m = lane & 15;
  v16h pack;
#pragma unroll
  for (int i = 0; i < 16; ++i) {
    int v = i >> 1, hh = i & 1;
    int kl = ((v >> 2) << 4) + (half << 3) + ((v & 3) << 1) + hh;
    pack[i] = (_Float16)A[(size_t)(mt * 16 + m) * ld + kt * 32 + kl];
  }
  ((v16h*)out)[(size_t)tile * 32 + lane] = pack;
}

// B (32x16 f16, from row-major W[N,K], B(k,n)=W[n,k]): lane = half*16 + n,
// slot i -> k_local = (i>>3)*16 + half*8 + (i&7).  Tile index = nt*Kt + kt.
__global__ void k_pack_b(const float* __restrict__ W, int ld, int koff,
                         int N, int K, _Float16* __restrict__ out) {
  int Kt = K >> 5;
  int tid = blockIdx.x * blockDim.x + threadIdx.x;
  int total = (N >> 4) * Kt * 32;
  if (tid >= total) return;
  int lane = tid & 31;
  int tile = tid >> 5;
  int kt = tile % Kt, nt = tile / Kt;
  v16h pack;
#pragma unroll
  for (int i = 0; i < 16; ++i)
    pack[i] = (_Float16)W[(size_t)(nt * 16 + (lane & 15)) * ld + koff + kt * 32 + ((i >> 3) << 4) + ((lane >> 4) << 3) + (i & 7)];
  ((v16h*)out)[(size_t)tile * 32 + lane] = pack;
}

// ---------------- WMMA GEMM: C[M,N] = A[M,K] * W[N,K]^T (+bias) ------------
// One wave computes a 64(M) x 32(N) block: 4 M-tiles x 2 N-tiles, so each
// packed B fragment is loaded from L2 by exactly one wave (B traffic cut 4x)
// and each k-step issues 8 WMMAs for 6 fragment loads.
__global__ void k_gemm_f16(const _Float16* __restrict__ pA,
                           const _Float16* __restrict__ pB,
                           const float* __restrict__ bias,
                           float* __restrict__ C, int M, int N, int K) {
  const int Kt = K >> 5;
  const int strips = N >> 5;                 // 2 N-tiles per strip
  const int totalWaves = (M >> 6) * strips;  // 4 M-tiles per wave
  const int wave = blockIdx.x * (blockDim.x >> 5) + (threadIdx.x >> 5);
  if (wave >= totalWaves) return;
  const int lane = threadIdx.x & 31;
  const int mtb = wave / strips;  // block of 4 consecutive M-tiles
  const int ns = wave % strips;   // block of 2 consecutive N-tiles

  const v16h* Av = (const v16h*)pA;
  const v16h* Bv = (const v16h*)pB;
  const int ncol = lane & 15;
  const int half = lane >> 4;

  v8f acc[4][2];
#pragma unroll
  for (int j = 0; j < 2; ++j) {
    float bz = bias ? bias[ns * 32 + j * 16 + ncol] : 0.0f;
#pragma unroll
    for (int i = 0; i < 4; ++i)
#pragma unroll
      for (int r = 0; r < 8; ++r) acc[i][j][r] = bz;
  }

  for (int kt = 0; kt < Kt; ++kt) {
    v16h af[4];
#pragma unroll
    for (int i = 0; i < 4; ++i)
      af[i] = Av[((size_t)(mtb * 4 + i) * Kt + kt) * 32 + lane];
    v16h bf[2];
#pragma unroll
    for (int j = 0; j < 2; ++j)
      bf[j] = Bv[((size_t)(ns * 2 + j) * Kt + kt) * 32 + lane];
    if (kt + 1 < Kt) {  // prefetch next k-step of the streamed (B) operand
      __builtin_prefetch(&Bv[((size_t)(ns * 2) * Kt + kt + 1) * 32 + lane], 0, 3);
      __builtin_prefetch(&Bv[((size_t)(ns * 2 + 1) * Kt + kt + 1) * 32 + lane], 0, 3);
    }
#pragma unroll
    for (int j = 0; j < 2; ++j)
#pragma unroll
      for (int i = 0; i < 4; ++i)
        acc[i][j] = __builtin_amdgcn_wmma_f32_16x16x32_f16(
            false, af[i], false, bf[j], (short)0, acc[i][j], false, false);
  }

#pragma unroll
  for (int i = 0; i < 4; ++i)
#pragma unroll
    for (int j = 0; j < 2; ++j)
#pragma unroll
      for (int r = 0; r < 8; ++r) {
        int m = mtb * 64 + i * 16 + r + half * 8;
        int n = ns * 32 + j * 16 + ncol;
        C[(size_t)m * N + n] = acc[i][j][r];
      }
}

// ---------------- elementwise / small kernels ------------------------------
__global__ void k_ffill(float* p, int n, float v) {
  int i = blockIdx.x * blockDim.x + threadIdx.x;
  if (i < n) p[i] = v;
}
__global__ void k_ifill(int* p, int n, int v) {
  int i = blockIdx.x * blockDim.x + threadIdx.x;
  if (i < n) p[i] = v;
}
__global__ void k_tanh_ip(float* p, int n) {
  int i = blockIdx.x * blockDim.x + threadIdx.x;
  if (i < n) p[i] = tanhf(p[i]);
}

__global__ void k_embed_gather(const int* __restrict__ idxs,
                               const float* __restrict__ emb,
                               float* __restrict__ out, int nRows, int ecols) {
  int i = blockIdx.x * blockDim.x + threadIdx.x;
  if (i >= nRows * ecols) return;
  int r = i / ecols, c = i - r * ecols;
  out[i] = emb[(size_t)idxs[r] * ecols + c];
}

// PyTorch GRU gate math; gi/gh already include their biases (GEMM epilogue).
// Writes h in place; optionally mirrors h into the encoder output sequence.
__global__ void k_gru_gate(const float* __restrict__ gi,
                           const float* __restrict__ gh,
                           float* __restrict__ h,
                           float* __restrict__ seqOut, int seqStride) {
  int idx = blockIdx.x * blockDim.x + threadIdx.x;
  if (idx >= BN * HH) return;
  int b = idx >> 9, j = idx & (HH - 1);
  float ir = gi[b * 3 * HH + j];
  float iz = gi[b * 3 * HH + HH + j];
  float in_ = gi[b * 3 * HH + 2 * HH + j];
  float hr = gh[b * 3 * HH + j];
  float hz = gh[b * 3 * HH + HH + j];
  float hn = gh[b * 3 * HH + 2 * HH + j];
  float r = 1.0f / (1.0f + __expf(-(ir + hr)));
  float z = 1.0f / (1.0f + __expf(-(iz + hz)));
  float n = tanhf(in_ + r * hn);
  float hv = (1.0f - z) * n + z * h[b * HH + j];
  h[b * HH + j] = hv;
  if (seqOut) seqOut[(size_t)b * seqStride + j] = hv;
}

// s[t,b] = tanh(Uh[t,b,:] + wh[b,:]) . w_a2 + b_a2   (one wave per (t,b))
__global__ void k_attn_score(const float* __restrict__ Uh,
                             const float* __restrict__ wh,
                             const float* __restrict__ wa2,
                             const float* __restrict__ ba2,
                             float* __restrict__ s) {
  int row = blockIdx.x * (blockDim.x >> 5) + (threadIdx.x >> 5);
  if (row >= TX * BN) return;
  int lane = threadIdx.x & 31;
  int b = row & (BN - 1);
  float acc = 0.0f;
  for (int j = lane; j < HH; j += 32)
    acc += tanhf(Uh[(size_t)row * HH + j] + wh[b * HH + j]) * wa2[j];
  for (int off = 16; off; off >>= 1) acc += __shfl_xor(acc, off, 32);
  if (lane == 0) s[row] = acc + ba2[0];
}

// softmax over t (axis 0) for each b; one 64-thread block per b
__global__ void k_attn_softmax(const float* __restrict__ s, float* __restrict__ w) {
  __shared__ float sm[TX];
  int b = blockIdx.x, t = threadIdx.x;
  float v = s[t * BN + b];
  sm[t] = v;
  __syncthreads();
  for (int off = 32; off; off >>= 1) {
    if (t < off) sm[t] = fmaxf(sm[t], sm[t + off]);
    __syncthreads();
  }
  float m = sm[0];
  __syncthreads();
  float e = __expf(v - m);
  sm[t] = e;
  __syncthreads();
  for (int off = 32; off; off >>= 1) {
    if (t < off) sm[t] += sm[t + off];
    __syncthreads();
  }
  w[t * BN + b] = e / sm[0];
}

// context[b,c] = sum_t w[t,b] * enc[t,b,c]
__global__ void k_context(const float* __restrict__ w, const float* __restrict__ enc,
                          float* __restrict__ ctx) {
  int idx = blockIdx.x * blockDim.x + threadIdx.x;
  if (idx >= BN * 2 * HH) return;
  int b = idx >> 10, c = idx & (2 * HH - 1);
  float acc = 0.0f;
  for (int t = 0; t < TX; ++t)
    acc += w[t * BN + b] * enc[((size_t)t * BN + b) * (2 * HH) + c];
  ctx[idx] = acc;
}

__global__ void k_build_rnnin(const int* __restrict__ tok,
                              const float* __restrict__ dec_emb,
                              const float* __restrict__ ctx,
                              float* __restrict__ out) {
  int i = blockIdx.x * blockDim.x + threadIdx.x;
  const int W = EE + 2 * HH;  // 1280
  if (i >= BN * W) return;
  int b = i / W, c = i - b * W;
  out[i] = (c < EE) ? dec_emb[(size_t)tok[b] * EE + c] : ctx[b * (2 * HH) + (c - EE)];
}

__global__ void k_build_lsmin(const float* __restrict__ h,
                              const float* __restrict__ ctx,
                              float* __restrict__ out) {
  int i = blockIdx.x * blockDim.x + threadIdx.x;
  const int W = 3 * HH;  // 1536
  if (i >= BN * W) return;
  int b = i / W, c = i - b * W;
  out[i] = (c < HH) ? h[b * HH + c] : ctx[b * (2 * HH) + (c - HH)];
}

// log_softmax over Ky per row + argmax feedback; one 256-thread block per b
__global__ void k_logsoftmax(const float* __restrict__ logits,
                             float* __restrict__ out, int* __restrict__ tok) {
  __shared__ float sv[256];
  __shared__ int si[256];
  int b = blockIdx.x, tid = threadIdx.x;
  const float* row = logits + (size_t)b * KY;
  float m = -3.4e38f;
  int mi = 0x7fffffff;
  for (int k = tid; k < KY; k += 256) {
    float v = row[k];
    if (v > m || (v == m && k < mi)) { m = v; mi = k; }
  }
  sv[tid] = m; si[tid] = mi;
  __syncthreads();
  for (int off = 128; off; off >>= 1) {
    if (tid < off) {
      if (sv[tid + off] > sv[tid] ||
          (sv[tid + off] == sv[tid] && si[tid + off] < si[tid])) {
        sv[tid] = sv[tid + off]; si[tid] = si[tid + off];
      }
    }
    __syncthreads();
  }
  float rm = sv[0];
  int ri = si[0];
  __syncthreads();
  float sum = 0.0f;
  for (int k = tid; k < KY; k += 256) sum += __expf(row[k] - rm);
  sv[tid] = sum;
  __syncthreads();
  for (int off = 128; off; off >>= 1) {
    if (tid < off) sv[tid] += sv[tid + off];
    __syncthreads();
  }
  float lse = rm + __logf(sv[0]);
  for (int k = tid; k < KY; k += 256) out[(size_t)b * KY + k] = row[k] - lse;
  if (tid == 0) tok[b] = ri;
}

// ---------------- host-side launch helpers ---------------------------------
static void gemm(hipStream_t s, const _Float16* pA, const _Float16* pB,
                 const float* bias, float* C, int M, int N, int K) {
  int waves = (M >> 6) * (N >> 5);
  k_gemm_f16<<<(waves + 3) / 4, 128, 0, s>>>(pA, pB, bias, C, M, N, K);
}
static void packA(hipStream_t s, const float* A, int ld, int M, int K, _Float16* o) {
  int total = (M >> 4) * (K >> 5) * 32;
  k_pack_a<<<(total + 255) / 256, 256, 0, s>>>(A, ld, M, K, o);
}
static void packB(hipStream_t s, const float* W, int ld, int koff, int N, int K,
                  _Float16* o) {
  int total = (N >> 4) * (K >> 5) * 32;
  k_pack_b<<<(total + 255) / 256, 256, 0, s>>>(W, ld, koff, N, K, o);
}

extern "C" void kernel_launch(void* const* d_in, const int* in_sizes, int n_in,
                              void* d_out, int out_size, void* d_ws, size_t ws_size,
                              hipStream_t stream) {
  (void)in_sizes; (void)n_in; (void)out_size; (void)ws_size;
  const int*   inputs  = (const int*)  d_in[0];
  const float* enc_emb = (const float*)d_in[2];
  const float* W_ih_f  = (const float*)d_in[3];
  const float* W_hh_f  = (const float*)d_in[4];
  const float* b_ih_f  = (const float*)d_in[5];
  const float* b_hh_f  = (const float*)d_in[6];
  const float* W_ih_b  = (const float*)d_in[7];
  const float* W_hh_b  = (const float*)d_in[8];
  const float* b_ih_b  = (const float*)d_in[9];
  const float* b_hh_b  = (const float*)d_in[10];
  const float* W_init  = (const float*)d_in[11];
  const float* b_init  = (const float*)d_in[12];
  const float* dec_emb = (const float*)d_in[13];
  const float* W_ih_d  = (const float*)d_in[14];
  const float* W_hh_d  = (const float*)d_in[15];
  const float* b_ih_d  = (const float*)d_in[16];
  const float* b_hh_d  = (const float*)d_in[17];
  const float* W_lsm   = (const float*)d_in[18];
  const float* b_lsm   = (const float*)d_in[19];
  const float* W_a1    = (const float*)d_in[20];
  const float* b_a1    = (const float*)d_in[21];
  const float* W_a2    = (const float*)d_in[22];
  const float* b_a2    = (const float*)d_in[23];
  float* out = (float*)d_out;

  // ---- workspace carve ----
  char* p = (char*)d_ws;
  auto alloc = [&](size_t bytes) -> void* {
    void* r = (void*)p;
    p += (bytes + 255) & ~(size_t)255;
    return r;
  };
  auto fh = [&](size_t n) { return (_Float16*)alloc(n * sizeof(_Float16)); };
  auto ff = [&](size_t n) { return (float*)alloc(n * sizeof(float)); };

  _Float16* pW_ihf  = fh((size_t)1536 * 256);
  _Float16* pW_hhf  = fh((size_t)1536 * 512);
  _Float16* pW_ihb  = fh((size_t)1536 * 256);
  _Float16* pW_hhb  = fh((size_t)1536 * 512);
  _Float16* pW_init = fh((size_t)512 * 512);
  _Float16* pW_ihd  = fh((size_t)1536 * 1280);
  _Float16* pW_hhd  = fh((size_t)1536 * 512);
  _Float16* pW_a1h  = fh((size_t)512 * 512);
  _Float16* pW_a1e  = fh((size_t)512 * 1024);
  _Float16* pW_lsm  = fh((size_t)KY * 1536);    // 98 MB: L2-resident
  _Float16* pAemb   = fh((size_t)TX * BN * 256);
  _Float16* pAenc   = fh((size_t)TX * BN * 1024);
  _Float16* pAh     = fh((size_t)BN * 512);
  _Float16* pArnn   = fh((size_t)BN * 1280);
  _Float16* pAlsm   = fh((size_t)BN * 1536);

  float* embeds  = ff((size_t)TX * BN * EE);
  float* gi_f    = ff((size_t)TX * BN * 1536);
  float* gi_b    = ff((size_t)TX * BN * 1536);
  float* enc_out = ff((size_t)TX * BN * 1024);
  float* Uh      = ff((size_t)TX * BN * 512);
  float* h_f     = ff((size_t)BN * 512);
  float* h_b     = ff((size_t)BN * 512);
  float* h       = ff((size_t)BN * 512);
  float* gh      = ff((size_t)BN * 1536);
  float* wh      = ff((size_t)BN * 512);
  float* sbuf    = ff((size_t)TX * BN);
  float* wbuf    = ff((size_t)TX * BN);
  float* ctx     = ff((size_t)BN * 1024);
  float* rnn_in  = ff((size_t)BN * 1280);
  float* gi_d    = ff((size_t)BN * 1536);
  float* gh_d    = ff((size_t)BN * 1536);
  float* lsm_in  = ff((size_t)BN * 1536);
  float* logits  = ff((size_t)BN * KY);
  int*   tok     = (int*)alloc(BN * sizeof(int));

  // ---- one-time weight packing (f32 -> f16 fragments) ----
  packB(stream, W_ih_f, 256, 0, 1536, 256, pW_ihf);
  packB(stream, W_hh_f, 512, 0, 1536, 512, pW_hhf);
  packB(stream, W_ih_b, 256, 0, 1536, 256, pW_ihb);
  packB(stream, W_hh_b, 512, 0, 1536, 512, pW_hhb);
  packB(stream, W_init, 512, 0, 512, 512, pW_init);
  packB(stream, W_ih_d, 1280, 0, 1536, 1280, pW_ihd);
  packB(stream, W_hh_d, 512, 0, 1536, 512, pW_hhd);
  packB(stream, W_a1, 1536, 0, 512, 512, pW_a1h);     // W_a1[:, :H]
  packB(stream, W_a1, 1536, 512, 512, 1024, pW_a1e);  // W_a1[:, H:]
  packB(stream, W_lsm, 1536, 0, KY, 1536, pW_lsm);

  // ---- encoder ----
  k_embed_gather<<<(TX * BN * EE + 255) / 256, 256, 0, stream>>>(
      inputs, enc_emb, embeds, TX * BN, EE);
  packA(stream, embeds, EE, TX * BN, EE, pAemb);
  // all-timestep input projections (two big GEMMs instead of 128 small ones)
  gemm(stream, pAemb, pW_ihf, b_ih_f, gi_f, TX * BN, 1536, 256);
  gemm(stream, pAemb, pW_ihb, b_ih_b, gi_b, TX * BN, 1536, 256);

  k_ffill<<<(BN * HH + 255) / 256, 256, 0, stream>>>(h_f, BN * HH, 0.0f);
  k_ffill<<<(BN * HH + 255) / 256, 256, 0, stream>>>(h_b, BN * HH, 0.0f);

  for (int t = 0; t < TX; ++t) {  // forward scan
    packA(stream, h_f, HH, BN, HH, pAh);
    gemm(stream, pAh, pW_hhf, b_hh_f, gh, BN, 1536, HH);
    k_gru_gate<<<(BN * HH + 255) / 256, 256, 0, stream>>>(
        gi_f + (size_t)t * BN * 1536, gh, h_f,
        enc_out + (size_t)t * BN * 1024, 1024);
  }
  for (int i = 0; i < TX; ++i) {  // backward scan (t = TX-1-i)
    int t = TX - 1 - i;
    packA(stream, h_b, HH, BN, HH, pAh);
    gemm(stream, pAh, pW_hhb, b_hh_b, gh, BN, 1536, HH);
    k_gru_gate<<<(BN * HH + 255) / 256, 256, 0, stream>>>(
        gi_b + (size_t)t * BN * 1536, gh, h_b,
        enc_out + (size_t)t * BN * 1024 + HH, 1024);
  }

  // decoder initial hidden = tanh(h_b @ W_init^T + b_init)
  packA(stream, h_b, HH, BN, HH, pAh);
  gemm(stream, pAh, pW_init, b_init, h, BN, 512, 512);
  k_tanh_ip<<<(BN * HH + 255) / 256, 256, 0, stream>>>(h, BN * HH);

  // Bahdanau precompute: Uh = enc_out @ W_a1[:,H:]^T + b_a1   (once)
  packA(stream, enc_out, 1024, TX * BN, 1024, pAenc);
  gemm(stream, pAenc, pW_a1e, b_a1, Uh, TX * BN, 512, 1024);

  k_ifill<<<1, BN, 0, stream>>>(tok, BN, 1);

  // ---- decoder loop ----
  for (int step = 0; step < TY; ++step) {
    packA(stream, h, HH, BN, HH, pAh);  // reused for wh and gh_d
    gemm(stream, pAh, pW_a1h, nullptr, wh, BN, 512, 512);
    k_attn_score<<<(TX * BN * 32 + 255) / 256, 256, 0, stream>>>(
        Uh, wh, W_a2, b_a2, sbuf);
    k_attn_softmax<<<BN, TX, 0, stream>>>(sbuf, wbuf);
    k_context<<<(BN * 2 * HH + 255) / 256, 256, 0, stream>>>(wbuf, enc_out, ctx);

    k_build_rnnin<<<(BN * 1280 + 255) / 256, 256, 0, stream>>>(
        tok, dec_emb, ctx, rnn_in);
    packA(stream, rnn_in, 1280, BN, 1280, pArnn);
    gemm(stream, pArnn, pW_ihd, b_ih_d, gi_d, BN, 1536, 1280);
    gemm(stream, pAh, pW_hhd, b_hh_d, gh_d, BN, 1536, HH);
    k_gru_gate<<<(BN * HH + 255) / 256, 256, 0, stream>>>(
        gi_d, gh_d, h, nullptr, 0);

    k_build_lsmin<<<(BN * 1536 + 255) / 256, 256, 0, stream>>>(h, ctx, lsm_in);
    packA(stream, lsm_in, 1536, BN, 1536, pAlsm);
    gemm(stream, pAlsm, pW_lsm, b_lsm, logits, BN, KY, 1536);  // dominant GEMM
    k_logsoftmax<<<BN, 256, 0, stream>>>(
        logits, out + (size_t)step * BN * KY, tok);
  }
}
